// Router_87462714016470
// MI455X (gfx1250) — matrix operand order repaired and checked by reading, not compiled, not run on hardware
//
#include <hip/hip_runtime.h>
#include <hip/hip_bf16.h>

// ---------------------------------------------------------------------------
// MoE router for MI455X (gfx1250, wave32).
//   raw = x @ gate_w^T   (fp32-faithful via bf16x3 split-precision WMMA)
//   logits = sigmoid(raw); z_loss = mean(raw^2)
//   grouped top-k expert selection + renormalized weights
// B operands (gate hi/lo bf16 planes) are double-buffered through LDS using
// the CDNA5 async global->LDS path (ASYNCcnt) when available.
// d_out layout: weights[T*8] | indices[T*8] | z_loss[1] | logits[T*E]
// ---------------------------------------------------------------------------

typedef __attribute__((ext_vector_type(16))) __bf16 v16bf;
typedef __attribute__((ext_vector_type(8)))  float  v8f;
typedef __attribute__((ext_vector_type(4)))  int    v4i;

constexpr int T_ = 16384;
constexpr int H_ = 4096;
constexpr int E_ = 256;
constexpr int BM = 32;                 // rows per block (2 wave-rows of 16)
constexpr int GEMM_BLOCKS = T_ / BM;   // 512
constexpr int LROW = 40;               // LDS row stride in bf16 (80B, conflict-free)

#if __has_builtin(__builtin_amdgcn_global_load_async_to_lds_b128)
#define ASYNC_COPY 1
#else
#define ASYNC_COPY 0
#endif

__device__ __forceinline__ void async_copy16(const __bf16* __restrict__ g,
                                             __bf16* l) {
#if ASYNC_COPY
  void* gv = (void*)g;   // drop const, stay generic
  void* lv = (void*)l;
  __builtin_amdgcn_global_load_async_to_lds_b128(
      (__attribute__((address_space(1))) v4i*)gv,
      (__attribute__((address_space(3))) v4i*)lv, 0, 0);
#else
  *(float4*)l = *(const float4*)g;   // synchronous fallback
#endif
}

__device__ __forceinline__ void wait_async0() {
#if ASYNC_COPY
#if __has_builtin(__builtin_amdgcn_s_wait_asynccnt)
  __builtin_amdgcn_s_wait_asynccnt(0);
#else
  asm volatile("s_wait_asynccnt 0" ::: "memory");
#endif
#endif
}

// Stage one 32-wide K chunk of both gate planes (256 x 32 bf16 each) into LDS.
// 16KB/plane as 1024 16B units; 256 threads x 4 units x 2 planes.
__device__ __forceinline__ void stage_chunk(const __bf16* __restrict__ gh,
                                            const __bf16* __restrict__ gl,
                                            __bf16* sh_, __bf16* sl_,
                                            int k, int tid) {
#pragma unroll
  for (int i = 0; i < 4; ++i) {
    const int u = tid + (i << 8);
    const int e = u >> 2;
    const int part = u & 3;
    const size_t go = (size_t)e * H_ + k + part * 8;
    const int lo_ = e * LROW + part * 8;
    async_copy16(gh + go, sh_ + lo_);
    async_copy16(gl + go, sl_ + lo_);
  }
}

// 32B LDS read as two explicit 16B loads (rows are 16B-aligned).
__device__ __forceinline__ v16bf load_v16bf_lds(const __bf16* p) {
  union { v16bf v; uint4 u[2]; } t;
  t.u[0] = *(const uint4*)(p);
  t.u[1] = *(const uint4*)(p + 8);
  return t.v;
}

// Split 8 consecutive fp32 into bf16 hi + bf16 lo (residual), RNE both times.
__device__ __forceinline__ void split8(const float4* __restrict__ p,
                                       v16bf& hi, v16bf& lo, int base) {
  float4 f0 = p[0];
  float4 f1 = p[1];
  float v[8] = {f0.x, f0.y, f0.z, f0.w, f1.x, f1.y, f1.z, f1.w};
#pragma unroll
  for (int i = 0; i < 8; ++i) {
    __bf16 h = (__bf16)v[i];
    hi[base + i] = h;
    lo[base + i] = (__bf16)(v[i] - (float)h);
  }
}

// ---------------------------------------------------------------------------
// Kernel 0: pre-split gate_w (E x H fp32) into bf16 hi/lo planes (L2-resident).
// ---------------------------------------------------------------------------
__global__ __launch_bounds__(256) void prep_gate_kernel(
    const float* __restrict__ gw, __bf16* __restrict__ gh, __bf16* __restrict__ gl) {
  int i = blockIdx.x * 256 + threadIdx.x;
  float f = gw[i];
  __bf16 h = (__bf16)f;
  gh[i] = h;
  gl[i] = (__bf16)(f - (float)h);
}

// ---------------------------------------------------------------------------
// Kernel 1: gate GEMM via v_wmma_f32_16x16x32_bf16 (bf16x3), sigmoid store,
// per-block sum(raw^2) partial for z_loss.
// Block: 256 thr = 8 waves laid out 2(M) x 4(N); wave tile = 16 x 64.
// B staged through LDS (async, double-buffered); A split on the fly.
// ---------------------------------------------------------------------------
__global__ __launch_bounds__(256) void gemm_router_kernel(
    const float* __restrict__ x,
    const __bf16* __restrict__ gh, const __bf16* __restrict__ gl,
    float* __restrict__ logits_out, float* __restrict__ partials) {
  __shared__ __align__(16) __bf16 sB[2][2][E_ * LROW];  // [buf][hi/lo] 80KB

  const int tid = threadIdx.x;
  const int wave = tid >> 5;
  const int lane = tid & 31;
  const int wm = wave >> 2;            // 0..1
  const int wn = wave & 3;             // 0..3
  const int rowBase = blockIdx.x * BM + wm * 16;
  const int colBase = wn * 64;         // 4 N-tiles of 16
  const int half = lane >> 4;          // K-half select per ISA 16-bit layouts
  const int mr = lane & 15;

  v8f acc[4] = {};

  const float* __restrict__ xrow = x + (size_t)(rowBase + mr) * H_;

  // Prologue: stage first K chunk.
  stage_chunk(gh, gl, sB[0][0], sB[0][1], 0, tid);
  wait_async0();
  __syncthreads();

  for (int k = 0; k < H_; k += 32) {
    const int buf = (k >> 5) & 1;
    if (k + 32 < H_)
      stage_chunk(gh, gl, sB[buf ^ 1][0], sB[buf ^ 1][1], k + 32, tid);

    // A tile: 16x32 bf16 layout -> lane holds K = {k+16h..+15}
    v16bf ahi, alo;
    split8((const float4*)(xrow + k + 16 * half), ahi, alo, 0);
    split8((const float4*)(xrow + k + 16 * half + 8), ahi, alo, 8);

    if (k + 128 < H_) __builtin_prefetch(xrow + k + 128, 0, 1);

#pragma unroll
    for (int t = 0; t < 4; ++t) {
      const int e = colBase + t * 16 + mr;        // B lane: N = lane&15
      const int boff = e * LROW + 16 * half;      // K = 16*half + j
      const v16bf Bh = load_v16bf_lds(&sB[buf][0][boff]);
      const v16bf Bl = load_v16bf_lds(&sB[buf][1][boff]);
      // D += Ahi*Bhi + Ahi*Blo + Alo*Bhi   (~fp32-accurate)
      acc[t] = __builtin_amdgcn_wmma_f32_16x16x32_bf16(
          false, ahi, false, Bh, (short)0, acc[t], false, false);
      acc[t] = __builtin_amdgcn_wmma_f32_16x16x32_bf16(
          false, ahi, false, Bl, (short)0, acc[t], false, false);
      acc[t] = __builtin_amdgcn_wmma_f32_16x16x32_bf16(
          false, alo, false, Bh, (short)0, acc[t], false, false);
    }

    wait_async0();      // this wave's staging of buf^1 complete
    __syncthreads();    // whole block: buf fully consumed, buf^1 fully staged
  }

  // Epilogue: sigmoid -> logits_out, accumulate sum(raw^2) for z_loss.
  float ss = 0.0f;
#pragma unroll
  for (int t = 0; t < 4; ++t) {
    const int col = colBase + t * 16 + mr;
#pragma unroll
    for (int i = 0; i < 8; ++i) {
      float r = acc[t][i];
      ss += r * r;
      const int row = rowBase + i + 8 * half;   // C/D layout: M = i + 8*half
      logits_out[(size_t)row * E_ + col] = 1.0f / (1.0f + __expf(-r));
    }
  }
#pragma unroll
  for (int off = 16; off > 0; off >>= 1) ss += __shfl_xor(ss, off, 32);

  __shared__ float ssh[8];
  if (lane == 0) ssh[wave] = ss;
  __syncthreads();
  if (tid == 0) {
    float tot = 0.0f;
#pragma unroll
    for (int w = 0; w < 8; ++w) tot += ssh[w];
    partials[blockIdx.x] = tot;          // deterministic (no float atomics)
  }
}

// ---------------------------------------------------------------------------
// Kernel 2: deterministic z_loss reduction of per-block partials.
// ---------------------------------------------------------------------------
__global__ __launch_bounds__(256) void zloss_reduce_kernel(
    const float* __restrict__ partials, int n, float* __restrict__ zout) {
  __shared__ float sh[256];
  float s = 0.0f;
  for (int i = threadIdx.x; i < n; i += 256) s += partials[i];
  sh[threadIdx.x] = s;
  __syncthreads();
  for (int o = 128; o > 0; o >>= 1) {
    if (threadIdx.x < o) sh[threadIdx.x] += sh[threadIdx.x + o];
    __syncthreads();
  }
  if (threadIdx.x == 0) *zout = sh[0] / ((float)T_ * (float)E_);
}

// ---------------------------------------------------------------------------
// Kernel 3: grouped top-k selection, one thread per token row.
// Strict '>' comparisons reproduce jax.lax.top_k lowest-index tie-breaking.
// ---------------------------------------------------------------------------
__global__ __launch_bounds__(256) void topk_kernel(
    const float* __restrict__ logits,       // sigmoid scores (T x E)
    const float* __restrict__ bias,         // correction_bias (E)
    float* __restrict__ w_out, float* __restrict__ i_out) {
  const int row = blockIdx.x * 256 + threadIdx.x;
  if (row >= T_) return;
  const float* __restrict__ lr = logits + (size_t)row * E_;
  const float NEG = -3.402823466e+38f;

  // group scores = top1 + top2 of (sigmoid + bias) within each group of 32
  float gs[8];
#pragma unroll
  for (int g = 0; g < 8; ++g) {
    float m1 = NEG, m2 = NEG;
    for (int j = 0; j < 32; ++j) {
      const int e = g * 32 + j;
      const float s = lr[e] + bias[e];
      if (s > m1) { m2 = m1; m1 = s; }
      else if (s > m2) { m2 = s; }
    }
    gs[g] = m1 + m2;
  }

  // top-4 groups (first occurrence wins ties -> lowest index)
  unsigned gmask = 0;
#pragma unroll
  for (int r = 0; r < 4; ++r) {
    int best = 0; float bv = NEG;
#pragma unroll
    for (int g = 0; g < 8; ++g)
      if (!((gmask >> g) & 1u) && gs[g] > bv) { bv = gs[g]; best = g; }
    gmask |= 1u << best;
  }

  // top-8 experts among surviving groups (descending insertion)
  float val[8];
  int   idx[8];
#pragma unroll
  for (int i = 0; i < 8; ++i) { val[i] = NEG; idx[i] = 0; }
  for (int g = 0; g < 8; ++g) {
    if (!((gmask >> g) & 1u)) continue;
    for (int j = 0; j < 32; ++j) {
      const int e = g * 32 + j;
      const float s = lr[e] + bias[e];
      if (s > val[7]) {
        int p = 7;
#pragma unroll
        while (p > 0 && s > val[p - 1]) { val[p] = val[p - 1]; idx[p] = idx[p - 1]; --p; }
        val[p] = s; idx[p] = e;
      }
    }
  }

  // combine weights from bias-free sigmoid scores, renormalized
  float w[8], sum = 0.0f;
#pragma unroll
  for (int i = 0; i < 8; ++i) { w[i] = lr[idx[i]]; sum += w[i]; }
  const float d = fmaxf(sum, 1e-9f);
#pragma unroll
  for (int i = 0; i < 8; ++i) {
    w_out[(size_t)row * 8 + i] = w[i] / d;
    i_out[(size_t)row * 8 + i] = (float)idx[i];
  }
}

// ---------------------------------------------------------------------------
extern "C" void kernel_launch(void* const* d_in, const int* in_sizes, int n_in,
                              void* d_out, int out_size, void* d_ws, size_t ws_size,
                              hipStream_t stream) {
  const float* x    = (const float*)d_in[0];   // (T,H)
  const float* gw   = (const float*)d_in[1];   // (E,H)
  const float* bias = (const float*)d_in[2];   // (E,)

  float* out  = (float*)d_out;
  float* wout = out;                        // weights: T*8
  float* iout = out + (size_t)T_ * 8;       // indices (as float): T*8
  float* zout = out + (size_t)T_ * 16;      // z_loss: 1
  float* lout = out + (size_t)T_ * 16 + 1;  // logits: T*E

  // workspace: bf16 hi plane | bf16 lo plane | z_loss partials
  char* ws = (char*)d_ws;
  __bf16* gh     = (__bf16*)ws;                                  // E*H*2 B
  __bf16* gl     = (__bf16*)(ws + (size_t)E_ * H_ * 2);          // E*H*2 B
  float* partial = (float*)(ws + (size_t)E_ * H_ * 4);           // 512 floats

  prep_gate_kernel<<<(E_ * H_) / 256, 256, 0, stream>>>(gw, gh, gl);
  gemm_router_kernel<<<GEMM_BLOCKS, 256, 0, stream>>>(x, gh, gl, lout, partial);
  zloss_reduce_kernel<<<1, 256, 0, stream>>>(partial, GEMM_BLOCKS, zout);
  topk_kernel<<<T_ / 256, 256, 0, stream>>>(lout, bias, wout, iout);
}
